// GCNII_9345848836278
// MI455X (gfx1250) — matrix operand order repaired and checked by reading, not compiled, not run on hardware
//
#include <hip/hip_runtime.h>
#include <hip/hip_bf16.h>
#include <math.h>

// ---------------------------------------------------------------------------
// Types for WMMA fragments (wave32, v_wmma_f32_16x16x32_bf16)
// ---------------------------------------------------------------------------
typedef __attribute__((ext_vector_type(16))) __bf16 v16bf;
typedef __attribute__((ext_vector_type(8)))  float  v8f;
typedef __attribute__((ext_vector_type(8)))  int    v8i;
typedef __attribute__((ext_vector_type(4)))  int    v4i;

union FragBf {
    unsigned u[8];
    v4i      h[2];
    v8i      i8;
    v16bf    b;
};

__device__ inline unsigned short f2bf(float f) {
    // round-to-nearest-even f32 -> bf16
    unsigned u = __float_as_uint(f);
    u += 0x7fffu + ((u >> 16) & 1u);
    return (unsigned short)(u >> 16);
}
__device__ inline unsigned pk(float a, float b) {
    return (unsigned)f2bf(a) | ((unsigned)f2bf(b) << 16);
}

// ---------------------------------------------------------------------------
// Weight pre-pack: W[K,N] f32 (row-major) -> per-fragment bf16 dwords.
// ISA B-matrix 32x16 (16-bit) layout: V_i, lanes 0-15 hold K=2i,2i+1 (i<4), 8+2i,9+2i (i>=4);
// lanes 16-31 hold K=8+2i,9+2i (i<4), 16+2i,17+2i (i>=4); column N = lane%16.
// Fragment (nt,kt) = 256 dwords; lane l owns dwords [l*8 .. l*8+7].
// ---------------------------------------------------------------------------
__global__ void pack_w_kernel(const float* __restrict__ W, unsigned* __restrict__ out,
                              int K, int Nn, int nkt, int nnt) {
    int total = nkt * nnt * 256;
    for (int t = blockIdx.x * blockDim.x + threadIdx.x; t < total;
         t += gridDim.x * blockDim.x) {
        int i    = t & 7;
        int lane = (t >> 3) & 31;
        int frag = t >> 8;
        int kt   = frag % nkt;
        int nt   = frag / nkt;
        int n    = nt * 16 + (lane & 15);
        int k    = kt * 32 + ((lane < 16) ? 0 : 8) + 2 * i + ((i < 4) ? 0 : 8);
        float lo = (n < Nn && k     < K) ? W[(size_t)k       * Nn + n] : 0.f;
        float hi = (n < Nn && k + 1 < K) ? W[(size_t)(k + 1) * Nn + n] : 0.f;
        out[t] = pk(lo, hi);
    }
}

// ---------------------------------------------------------------------------
// fc0: X = relu(features[N,512] @ W0[512,128] + b0)
// One 16-row tile/block, 8 waves = 8 N-tiles, 16 K-tiles of WMMA.
// A staged in LDS as packed bf16 (ISA A-matrix 16x32 16-bit layout:
// M=lane%16; lanes<16: K 0-7 & 16-23; lanes>=16: K 8-15 & 24-31).
// ---------------------------------------------------------------------------
__global__ void __launch_bounds__(256) fc0_kernel(const float* __restrict__ F,
                                                  const unsigned* __restrict__ Bw,
                                                  const float* __restrict__ bias,
                                                  float* __restrict__ X, int Nrows) {
    __shared__ __align__(16) unsigned ldsA[16 * 256]; // 16 rows x 512 bf16
    const int  m0   = blockIdx.x * 16;
    const int  t    = threadIdx.x;
    const bool full = (m0 + 16 <= Nrows);

    // stage: 16 rows x 512 f32 -> bf16 pairs; float4 global loads
    #pragma unroll 2
    for (int j = 0; j < 8; ++j) {
        int rr  = j * 2 + (t >> 7);   // 0..15
        int k0  = (t & 127) * 4;      // 0..508
        int row = m0 + rr;
        float4 f = make_float4(0.f, 0.f, 0.f, 0.f);
        if (full || row < Nrows)
            f = *(const float4*)(F + (size_t)row * 512 + k0);
        ldsA[rr * 256 + (k0 >> 1)]     = pk(f.x, f.y);
        ldsA[rr * 256 + (k0 >> 1) + 1] = pk(f.z, f.w);
    }
    __syncthreads();

    const int wave = t >> 5, lane = t & 31;
    const int m    = lane & 15;
    v8f acc = {};
    #pragma unroll 4
    for (int kt = 0; kt < 16; ++kt) {
        int kb = kt * 16 + ((lane < 16) ? 0 : 4); // dword offset within row
        FragBf a, b;
        a.h[0] = *(const v4i*)&ldsA[m * 256 + kb];
        a.h[1] = *(const v4i*)&ldsA[m * 256 + kb + 8];
        b.i8   = *(const v8i*)(Bw + ((size_t)(wave * 16 + kt)) * 256 + lane * 8);
        acc = __builtin_amdgcn_wmma_f32_16x16x32_bf16(false, a.b, false, b.b,
                                                      (short)0, acc, false, false);
    }

    const int   ncol  = wave * 16 + (lane & 15);
    const float bv    = bias[ncol];
    const int   mbase = m0 + ((lane < 16) ? 0 : 8);
    if (full) {
        #pragma unroll
        for (int v = 0; v < 8; ++v) {
            float r = acc[v] + bv;
            X[(size_t)(mbase + v) * 128 + ncol] = r > 0.f ? r : 0.f;
        }
    } else {
        #pragma unroll
        for (int v = 0; v < 8; ++v) {
            int row = mbase + v;
            if (row < Nrows) {
                float r = acc[v] + bv;
                X[(size_t)row * 128 + ncol] = r > 0.f ? r : 0.f;
            }
        }
    }
}

// ---------------------------------------------------------------------------
// SpMM: agg[dst] += w_e * x[src]; one wave per edge (grid-stride),
// float4 row gather + next-edge prefetch, global_atomic_add_f32 scatter.
// x rows (25.6 MB) are L2-resident on the 192 MB L2.
// ---------------------------------------------------------------------------
__global__ void __launch_bounds__(256) spmm_kernel(const int* __restrict__ src,
                                                   const int* __restrict__ dst,
                                                   const float* __restrict__ w,
                                                   const float* __restrict__ x,
                                                   float* __restrict__ agg, int E) {
    const int nw   = (gridDim.x * blockDim.x) >> 5;
    const int wid  = (blockIdx.x * blockDim.x + threadIdx.x) >> 5;
    const int lane = threadIdx.x & 31;
    for (int e = wid; e < E; e += nw) {
        int en = e + nw;
        if (en < E) {   // pull next gather row toward the WGP (global_prefetch_b8)
            int sn = src[en];
            __builtin_prefetch(x + (size_t)sn * 128 + lane * 4, 0, 3);
        }
        int   s  = src[e];
        int   d  = dst[e];
        float we = w[e];
        float4 v = ((const float4*)(x + (size_t)s * 128))[lane];
        float* o = agg + (size_t)d * 128 + lane * 4;
        __hip_atomic_fetch_add(o + 0, we * v.x, __ATOMIC_RELAXED, __HIP_MEMORY_SCOPE_AGENT);
        __hip_atomic_fetch_add(o + 1, we * v.y, __ATOMIC_RELAXED, __HIP_MEMORY_SCOPE_AGENT);
        __hip_atomic_fetch_add(o + 2, we * v.z, __ATOMIC_RELAXED, __HIP_MEMORY_SCOPE_AGENT);
        __hip_atomic_fetch_add(o + 3, we * v.w, __ATOMIC_RELAXED, __HIP_MEMORY_SCOPE_AGENT);
    }
}

__global__ void zero_kernel(float4* __restrict__ p, int n4) {
    float4 z = make_float4(0.f, 0.f, 0.f, 0.f);
    for (int i = blockIdx.x * blockDim.x + threadIdx.x; i < n4;
         i += gridDim.x * blockDim.x)
        p[i] = z;
}

// ---------------------------------------------------------------------------
// GCNII layer GEMM: support = 0.9*agg + 0.1*h0 staged ONCE into LDS as both
// f32 (for the residual epilogue) and packed bf16 (for WMMA A-fragments);
// x = relu(beta*(support@W) + (1-beta)*support).
// ---------------------------------------------------------------------------
struct ConvSh {
    float    supf[16 * 128]; // 8 KB
    unsigned supb[16 * 64];  // 4 KB packed bf16
};

__global__ void __launch_bounds__(256) conv_kernel(const float* __restrict__ AGG,
                                                   const float* __restrict__ H0,
                                                   const unsigned* __restrict__ Bw,
                                                   float beta, float* __restrict__ X,
                                                   int Nrows) {
    __shared__ __align__(16) ConvSh sh;
    const int  m0   = blockIdx.x * 16;
    const int  t    = threadIdx.x;
    const bool full = (m0 + 16 <= Nrows);

    // stage: thread t owns 8 consecutive elements of row (t*8)>>7
    {
        const int e0  = t * 8;
        const int m   = e0 >> 7;
        const int k0  = e0 & 127;
        const int row = m0 + m;
        float s[8];
        if (full || row < Nrows) {
            const float4* ap = (const float4*)(AGG + (size_t)row * 128 + k0);
            const float4* hp = (const float4*)(H0  + (size_t)row * 128 + k0);
            float4 a0 = ap[0], a1 = ap[1];
            float4 h0 = hp[0], h1 = hp[1];
            s[0] = 0.9f * a0.x + 0.1f * h0.x;  s[1] = 0.9f * a0.y + 0.1f * h0.y;
            s[2] = 0.9f * a0.z + 0.1f * h0.z;  s[3] = 0.9f * a0.w + 0.1f * h0.w;
            s[4] = 0.9f * a1.x + 0.1f * h1.x;  s[5] = 0.9f * a1.y + 0.1f * h1.y;
            s[6] = 0.9f * a1.z + 0.1f * h1.z;  s[7] = 0.9f * a1.w + 0.1f * h1.w;
        } else {
            #pragma unroll
            for (int i = 0; i < 8; ++i) s[i] = 0.f;
        }
        float4* sf = (float4*)&sh.supf[e0];
        sf[0] = make_float4(s[0], s[1], s[2], s[3]);
        sf[1] = make_float4(s[4], s[5], s[6], s[7]);
        v4i pb = { (int)pk(s[0], s[1]), (int)pk(s[2], s[3]),
                   (int)pk(s[4], s[5]), (int)pk(s[6], s[7]) };
        *(v4i*)&sh.supb[e0 >> 1] = pb;
    }
    __syncthreads();

    const int wave = t >> 5, lane = t & 31;
    const int m    = lane & 15;
    v8f acc = {};
    #pragma unroll
    for (int kt = 0; kt < 4; ++kt) {
        int kb = kt * 16 + ((lane < 16) ? 0 : 4);
        FragBf a, b;
        a.h[0] = *(const v4i*)&sh.supb[m * 64 + kb];
        a.h[1] = *(const v4i*)&sh.supb[m * 64 + kb + 8];
        b.i8   = *(const v8i*)(Bw + ((size_t)(wave * 4 + kt)) * 256 + lane * 8);
        acc = __builtin_amdgcn_wmma_f32_16x16x32_bf16(false, a.b, false, b.b,
                                                      (short)0, acc, false, false);
    }

    const int   ncol = wave * 16 + (lane & 15);
    const int   moff = (lane < 16) ? 0 : 8;
    const float ob   = 1.f - beta;
    if (full) {
        #pragma unroll
        for (int v = 0; v < 8; ++v) {
            float s = sh.supf[(moff + v) * 128 + ncol];
            float r = beta * acc[v] + ob * s;
            X[(size_t)(m0 + moff + v) * 128 + ncol] = r > 0.f ? r : 0.f;
        }
    } else {
        #pragma unroll
        for (int v = 0; v < 8; ++v) {
            int row = m0 + moff + v;
            if (row < Nrows) {
                float s = sh.supf[(moff + v) * 128 + ncol];
                float r = beta * acc[v] + ob * s;
                X[(size_t)row * 128 + ncol] = r > 0.f ? r : 0.f;
            }
        }
    }
}

// ---------------------------------------------------------------------------
// fc1: out = x @ W1[128,40] + b1  (N padded to 48 -> 3 N-tiles; 4 waves/block,
// wave 3 only helps stage). The x tile is a pure 16x128 f32 copy, so it is
// staged with GLOBAL_LOAD_ASYNC_TO_LDS_B128 (ASYNCcnt-tracked DMA into LDS),
// then converted in-LDS to packed bf16 fragments.
// ---------------------------------------------------------------------------
struct Fc1Sh {
    float    xa[16 * 128]; // 8 KB (async destination)
    unsigned xb[16 * 64];  // 4 KB packed bf16
};

__global__ void __launch_bounds__(128) fc1_kernel(const float* __restrict__ Xin,
                                                  const unsigned* __restrict__ Bw,
                                                  const float* __restrict__ bias,
                                                  float* __restrict__ Out, int Nrows) {
    __shared__ __align__(16) Fc1Sh sh;
    const int  m0   = blockIdx.x * 16;
    const int  t    = threadIdx.x;
    const bool full = (m0 + 16 <= Nrows);

    if (full) {
        // async copy: 2048 f32 = 512 x b128; 128 threads x 4 issues
        #pragma unroll
        for (int j = 0; j < 4; ++j) {
            int e = (j * 128 + t) * 4;
            // generic LDS pointers carry the LDS byte offset in the low 32 bits
            unsigned     lds = (unsigned)(unsigned long long)&sh.xa[e];
            const float* gp  = Xin + (size_t)m0 * 128 + e;
            asm volatile("global_load_async_to_lds_b128 %0, %1, off"
                         :: "v"(lds), "v"(gp) : "memory");
        }
        asm volatile("s_wait_asynccnt 0" ::: "memory");
    } else {
        #pragma unroll
        for (int j = 0; j < 16; ++j) {
            int e = j * 128 + t;
            int row = m0 + (e >> 7);
            sh.xa[e] = (row < Nrows) ? Xin[(size_t)row * 128 + (e & 127)] : 0.f;
        }
    }
    __syncthreads();

    // in-LDS f32 -> packed bf16: thread t owns 16 consecutive elements
    {
        const int e0 = t * 16;
        const float4* xf = (const float4*)&sh.xa[e0];
        float4 f0 = xf[0], f1 = xf[1], f2 = xf[2], f3 = xf[3];
        v4i p0 = { (int)pk(f0.x, f0.y), (int)pk(f0.z, f0.w),
                   (int)pk(f1.x, f1.y), (int)pk(f1.z, f1.w) };
        v4i p1 = { (int)pk(f2.x, f2.y), (int)pk(f2.z, f2.w),
                   (int)pk(f3.x, f3.y), (int)pk(f3.z, f3.w) };
        *(v4i*)&sh.xb[(e0 >> 1)]     = p0;
        *(v4i*)&sh.xb[(e0 >> 1) + 4] = p1;
    }
    __syncthreads();

    const int wave = t >> 5, lane = t & 31;
    if (wave < 3) {
        const int m = lane & 15;
        v8f acc = {};
        #pragma unroll
        for (int kt = 0; kt < 4; ++kt) {
            int kb = kt * 16 + ((lane < 16) ? 0 : 4);
            FragBf a, b;
            a.h[0] = *(const v4i*)&sh.xb[m * 64 + kb];
            a.h[1] = *(const v4i*)&sh.xb[m * 64 + kb + 8];
            b.i8   = *(const v8i*)(Bw + ((size_t)(wave * 4 + kt)) * 256 + lane * 8);
            acc = __builtin_amdgcn_wmma_f32_16x16x32_bf16(false, a.b, false, b.b,
                                                          (short)0, acc, false, false);
        }
        const int nc = wave * 16 + (lane & 15);
        if (nc < 40) {
            const float bv   = bias[nc];
            const int   moff = (lane < 16) ? 0 : 8;
            if (full) {
                #pragma unroll
                for (int v = 0; v < 8; ++v)
                    Out[(size_t)(m0 + moff + v) * 40 + nc] = acc[v] + bv;
            } else {
                #pragma unroll
                for (int v = 0; v < 8; ++v) {
                    int row = m0 + moff + v;
                    if (row < Nrows)
                        Out[(size_t)row * 40 + nc] = acc[v] + bv;
                }
            }
        }
    }
}

// ---------------------------------------------------------------------------
// Host orchestration
// ---------------------------------------------------------------------------
extern "C" void kernel_launch(void* const* d_in, const int* in_sizes, int n_in,
                              void* d_out, int out_size, void* d_ws, size_t ws_size,
                              hipStream_t stream) {
    const float* features = (const float*)d_in[0];
    const int*   edge_idx = (const int*)d_in[1];
    const float* norm_A   = (const float*)d_in[2];
    const float* w_fc0    = (const float*)d_in[3];
    const float* b_fc0    = (const float*)d_in[4];
    const float* conv_w   = (const float*)d_in[5];
    const float* w_fc1    = (const float*)d_in[6];
    const float* b_fc1    = (const float*)d_in[7];
    float*       out      = (float*)d_out;

    const int Nrows = in_sizes[0] / 512;
    const int E     = in_sizes[1] / 2;
    const int* src  = edge_idx;
    const int* dst  = edge_idx + E;

    // workspace carve-out (256B aligned)
    char*  ws = (char*)d_ws;
    size_t o  = 0;
    auto carve = [&](size_t bytes) -> void* {
        void* p = ws + o;
        o = (o + bytes + 255) & ~(size_t)255;
        return p;
    };
    float*    h0      = (float*)carve((size_t)Nrows * 128 * 4);
    float*    x       = (float*)carve((size_t)Nrows * 128 * 4);
    float*    agg     = (float*)carve((size_t)Nrows * 128 * 4);
    unsigned* pk_fc0  = (unsigned*)carve((size_t)16 * 8 * 256 * 4);
    unsigned* pk_conv = (unsigned*)carve((size_t)8 * 4 * 8 * 256 * 4);
    unsigned* pk_fc1  = (unsigned*)carve((size_t)4 * 3 * 256 * 4);

    // 1) pack weights into bf16 WMMA fragment layout
    pack_w_kernel<<<128, 256, 0, stream>>>(w_fc0, pk_fc0, 512, 128, 16, 8);
    for (int i = 0; i < 8; ++i)
        pack_w_kernel<<<32, 256, 0, stream>>>(conv_w + (size_t)i * 128 * 128,
                                              pk_conv + (size_t)i * 8192, 128, 128, 4, 8);
    pack_w_kernel<<<12, 256, 0, stream>>>(w_fc1, pk_fc1, 128, 40, 4, 3);

    const int mt = (Nrows + 15) / 16;

    // 2) fc0 -> h0
    fc0_kernel<<<mt, 256, 0, stream>>>(features, pk_fc0, b_fc0, h0, Nrows);

    // 3) 8 GCNII layers
    const int n4 = Nrows * 128 / 4;
    for (int i = 0; i < 8; ++i) {
        float beta = logf(0.5f / (float)(i + 1) + 1.0f);
        zero_kernel<<<1024, 256, 0, stream>>>((float4*)agg, n4);
        const float* xin = (i == 0) ? h0 : x;
        spmm_kernel<<<4096, 256, 0, stream>>>(src, dst, norm_A, xin, agg, E);
        conv_kernel<<<mt, 256, 0, stream>>>(agg, h0, pk_conv + (size_t)i * 8192,
                                            beta, x, Nrows);
    }

    // 4) fc1 -> out
    fc1_kernel<<<mt, 128, 0, stream>>>(x, pk_fc1, b_fc1, out, Nrows);
}
